// SpreadsheetTableFinder_26199300505770
// MI455X (gfx1250) — compile-verified, matrix operand order checked by reading, not run on hardware
//
#include <hip/hip_runtime.h>
#include <hip/hip_bf16.h>

typedef _Float16 half8  __attribute__((ext_vector_type(8)));
typedef _Float16 v16h   __attribute__((ext_vector_type(16)));
typedef float    v8f    __attribute__((ext_vector_type(8)));

#define B_      4
#define IMG_    256
#define H1      128     // conv1 output spatial
#define C1      64      // conv1 channels
#define H2      64      // conv2 output spatial
#define C2      256     // conv2 channels
#define NPROP   1000
#define NGT     20
#define SAMPLES_ 256
#define NROI    512     // sampled props + matched gts
#define OUTSZ   7
#define KDIM    12544   // 256*7*7
#define KCH     (KDIM/32)   // 392 K-chunks for the FC WMMA
#define K2      576     // conv2 K = 9*64
#define ACHUNKS (16 * K2 / 8)   // 1152 b128 blocks to stage one A tile

// K index inside a 32-wide chunk for f16 A/B fragments (ISA 7.12.2):
// elements 0..7 -> K = g*8 + e ; elements 8..15 -> K = 16 + g*8 + (e-8)
__device__ __forceinline__ int kmap(int e, int g) {
    return (e < 8) ? (g * 8 + e) : (8 + g * 8 + e);
}

// ---------------------------------------------------------------------------
// conv1: 3->64, 3x3, stride 2, SAME (pad hi only), ReLU.
// Output stored NHWC f16 so it is the contiguous-K im2col operand for conv2.
// ---------------------------------------------------------------------------
__global__ void conv1_kernel(const float* __restrict__ img,
                             const float* __restrict__ w,
                             const float* __restrict__ bias,
                             _Float16* __restrict__ f1h) {
    int t  = blockIdx.x * blockDim.x + threadIdx.x;      // 4*64*128*128 = 2^22
    int ox = t & 127;
    int oy = (t >> 7) & 127;
    int co = (t >> 14) & 63;
    int b  = t >> 20;
    float acc = bias[co];
    #pragma unroll
    for (int ci = 0; ci < 3; ++ci)
        #pragma unroll
        for (int ky = 0; ky < 3; ++ky) {
            int iy = oy * 2 + ky;
            if (iy >= IMG_) continue;
            #pragma unroll
            for (int kx = 0; kx < 3; ++kx) {
                int ix = ox * 2 + kx;
                if (ix >= IMG_) continue;
                acc += img[((b * 3 + ci) * IMG_ + iy) * IMG_ + ix] *
                       w[((co * 3 + ci) * 3 + ky) * 3 + kx];
            }
        }
    acc = fmaxf(acc, 0.f);
    f1h[((size_t)((b * H1 + oy) * H1 + ox)) * C1 + co] = (_Float16)acc;
}

// ---------------------------------------------------------------------------
// Pre-swizzle conv2 weights (OIHW f32) into per-lane WMMA B fragments:
// wfrag2[((chunk*16 + ntile)*32 + lane)*16 + e]
// ---------------------------------------------------------------------------
__global__ void pack2_kernel(const float* __restrict__ w2,
                             _Float16* __restrict__ wfrag2) {
    int t = blockIdx.x * blockDim.x + threadIdx.x;       // 18*16*32*16 = 147456
    if (t >= 18 * 16 * 32 * 16) return;
    int e  = t & 15;
    int L  = (t >> 4) & 31;
    int nt = (t >> 9) & 15;
    int ch = t >> 13;                                    // 0..17
    int g  = L >> 4;
    int K  = ch * 32 + kmap(e, g);                       // 0..575
    int tap = K >> 6, cin = K & 63;
    int ky = tap / 3, kx = tap % 3;
    int co = nt * 16 + (L & 15);
    wfrag2[t] = (_Float16)w2[((co * C1 + cin) * 3 + ky) * 3 + kx];
}

// ---------------------------------------------------------------------------
// conv2 as implicit GEMM via v_wmma_f32_16x16x32_f16 with async-to-LDS
// staging of the shared A (im2col) tile.
// Block = 256 threads = 8 waves; block owns one 16-pixel M-tile and an
// 8-wide N-tile group; A tile (16 x 576 f16 = 18KB) staged once per block
// via global_load_async_to_lds_b128, reused by all 8 waves from LDS.
// ---------------------------------------------------------------------------
__global__ void conv2_wmma_kernel(const _Float16* __restrict__ f1h,
                                  const _Float16* __restrict__ wfrag2,
                                  const float* __restrict__ bias2,
                                  float* __restrict__ f2) {
    __shared__ __align__(16) _Float16 sA[16 * K2];       // [m][K], K contiguous

    int ntg = blockIdx.x & 1;                            // N-tile group (0..1)
    int mt  = (blockIdx.x >> 1) & 255;                   // M tile (0..255)
    int b   = blockIdx.x >> 9;                           // batch
    int tid = threadIdx.x;

    // ---- stage A tile: 1152 x 16B blocks; async copy or zero-fill (SAME pad)
    unsigned lds_base = (unsigned)(size_t)(&sA[0]);
    for (int idx = tid; idx < ACHUNKS; idx += 256) {
        int m  = idx / (K2 / 8);                         // pixel row 0..15
        int r8 = idx % (K2 / 8);                         // 8-half block in K
        int Kb = r8 * 8;
        int tap = Kb >> 6, c = Kb & 63;                  // stays inside one tap
        int ky = tap / 3, kx = tap % 3;
        int p  = mt * 16 + m;
        int oy = p >> 6, ox = p & 63;
        int iy = oy * 2 + ky, ix = ox * 2 + kx;          // SAME: pad high side
        if (iy < H1 && ix < H1) {
            unsigned lds_off = lds_base + (unsigned)(idx * 16);
            unsigned long long gaddr = (unsigned long long)(size_t)
                (f1h + ((size_t)((b * H1 + iy) * H1 + ix)) * C1 + c);
            asm volatile("global_load_async_to_lds_b128 %0, %1, off"
                         :: "v"(lds_off), "v"(gaddr) : "memory");
        } else {
            half8 z = {};
            *(half8*)(&sA[idx * 8]) = z;                 // compiler-visible zero
        }
    }
    asm volatile("s_wait_asynccnt 0x0" ::: "memory");
    __syncthreads();

    // ---- WMMA main loop: each wave does one 16x16 output tile
    int lane = tid & 31;
    int nt   = ntg * 8 + (tid >> 5);                     // N tile 0..15
    int g    = lane >> 4;
    int m    = lane & 15;

    v8f acc = {};
    #pragma unroll
    for (int ch = 0; ch < 18; ++ch) {
        int base = m * K2 + ch * 32;
        half8 lo = *(const half8*)(&sA[base + g * 8]);
        half8 hi = *(const half8*)(&sA[base + 16 + g * 8]);
        v16h a = __builtin_shufflevector(lo, hi, 0,1,2,3,4,5,6,7,8,9,10,11,12,13,14,15);
        v16h bf = *(const v16h*)(wfrag2 + ((size_t)(ch * 16 + nt) * 32 + lane) * 16);
        acc = __builtin_amdgcn_wmma_f32_16x16x32_f16(
            false, a, false, bf, (short)0, acc, false, false);
    }
    // D layout: N = lane&15, M = v + 8*g
    int co = nt * 16 + m;
    float bv = bias2[co];
    #pragma unroll
    for (int v = 0; v < 8; ++v) {
        int ml = v + 8 * g;
        int pp = mt * 16 + ml;
        int oyy = pp >> 6, oxx = pp & 63;
        f2[(((size_t)b * C2 + co) * H2 + oyy) * H2 + oxx] = fmaxf(acc[v] + bv, 0.f);
    }
}

// ---------------------------------------------------------------------------
// IoU match + deterministic top-256 sampling (scores are distinct tiers).
// One block per batch.
// ---------------------------------------------------------------------------
__global__ void match_kernel(const float* __restrict__ gt,
                             const float* __restrict__ props,
                             float* __restrict__ rois) {
    int b = blockIdx.x, tid = threadIdx.x;
    __shared__ float sgt[NGT * 4];
    __shared__ int   spri[NPROP];
    __shared__ int   smat[NPROP];
    __shared__ int   scnt[3];
    __shared__ int   ssel[SAMPLES_];

    if (tid < NGT * 4) sgt[tid] = gt[b * NGT * 4 + tid];
    if (tid < 3) scnt[tid] = 0;
    __syncthreads();

    for (int i = tid; i < NPROP; i += blockDim.x) {
        const float* pb = props + (b * NPROP + i) * 4;
        float px1 = pb[0], py1 = pb[1], px2 = pb[2], py2 = pb[3];
        float pa = (px2 - px1) * (py2 - py1);
        float best = -1.f; int bi = 0;
        for (int j = 0; j < NGT; ++j) {
            float gx1 = sgt[j*4+0], gy1 = sgt[j*4+1], gx2 = sgt[j*4+2], gy2 = sgt[j*4+3];
            float ga = (gx2 - gx1) * (gy2 - gy1);
            float iw = fmaxf(fminf(px2, gx2) - fmaxf(px1, gx1), 0.f);
            float ih = fmaxf(fminf(py2, gy2) - fmaxf(py1, gy1), 0.f);
            float inter = iw * ih;
            float iou = inter / (pa + ga - inter);
            if (iou > best) { best = iou; bi = j; }
        }
        int matched = (best < 0.3f) ? -1 : ((best < 0.7f) ? -2 : bi);
        int pri = (matched >= 0) ? 2 : ((matched == -1) ? 1 : 0);
        spri[i] = pri; smat[i] = matched;
        atomicAdd(&scnt[pri], 1);
    }
    __syncthreads();
    if (tid == 0) {
        // rank = (#higher tier) + (#same tier with smaller index) == top_k order
        int base[3];
        base[2] = 0; base[1] = scnt[2]; base[0] = scnt[2] + scnt[1];
        for (int i = 0; i < NPROP; ++i) {
            int r = base[spri[i]]++;
            if (r < SAMPLES_) ssel[r] = i;
        }
    }
    __syncthreads();
    if (tid < SAMPLES_) {
        int i = ssel[tid];
        #pragma unroll
        for (int k = 0; k < 4; ++k)
            rois[((size_t)b * NROI + tid) * 4 + k] = props[(b * NPROP + i) * 4 + k];
        int mi = smat[i]; if (mi < 0) mi = 0;
        #pragma unroll
        for (int k = 0; k < 4; ++k)
            rois[((size_t)b * NROI + SAMPLES_ + tid) * 4 + k] = sgt[mi * 4 + k];
    }
}

// ---------------------------------------------------------------------------
// ROI-align (14x14 samples -> 7x7 avg). Writes x matrix rows directly as f16:
// xmat[row = b*512+r][K = c*49 + py*7 + px]
// ---------------------------------------------------------------------------
__global__ void roi_kernel(const float* __restrict__ f2,
                           const float* __restrict__ rois,
                           _Float16* __restrict__ xmat) {
    int t = blockIdx.x * blockDim.x + threadIdx.x;       // 4*512*256*49
    int px = t % 7;
    int py = (t / 7) % 7;
    int c  = (t / 49) % C2;
    int r  = (t / (49 * C2)) % NROI;
    int b  = t / (49 * C2 * NROI);
    const float* bx = rois + ((size_t)b * NROI + r) * 4;
    float x1 = bx[0] * 0.25f, y1 = bx[1] * 0.25f;
    float x2 = bx[2] * 0.25f, y2 = bx[3] * 0.25f;
    float rw = fmaxf(x2 - x1, 1.f), rh = fmaxf(y2 - y1, 1.f);
    const float* feat = f2 + ((size_t)(b * C2 + c)) * H2 * H2;
    float acc = 0.f;
    #pragma unroll
    for (int sy = 0; sy < 2; ++sy) {
        float gy = ((float)(2 * py + sy) + 0.5f) / 14.f;
        float yy = fminf(fmaxf(y1 + gy * rh, 0.f), 63.f);
        int y0 = (int)floorf(yy);
        int y1i = min(y0 + 1, 63);
        float wy = yy - (float)y0;
        #pragma unroll
        for (int sx = 0; sx < 2; ++sx) {
            float gx = ((float)(2 * px + sx) + 0.5f) / 14.f;
            float xx = fminf(fmaxf(x1 + gx * rw, 0.f), 63.f);
            int x0 = (int)floorf(xx);
            int x1i = min(x0 + 1, 63);
            float wx = xx - (float)x0;
            float v00 = feat[y0 * 64 + x0],  v01 = feat[y0 * 64 + x1i];
            float v10 = feat[y1i * 64 + x0], v11 = feat[y1i * 64 + x1i];
            acc += v00 * (1.f - wy) * (1.f - wx) + v01 * (1.f - wy) * wx +
                   v10 * wy * (1.f - wx)         + v11 * wy * wx;
        }
    }
    acc *= 0.25f;
    xmat[((size_t)(b * NROI + r)) * KDIM + c * 49 + py * 7 + px] = (_Float16)acc;
}

// ---------------------------------------------------------------------------
// Pack head weights into one 16-wide N tile of WMMA B fragments:
// n<2: cls_w row n ; n<10: bbox_w row n-2 ; else 0.
// ---------------------------------------------------------------------------
__global__ void packF_kernel(const float* __restrict__ clsw,
                             const float* __restrict__ bboxw,
                             _Float16* __restrict__ wfragF) {
    int t = blockIdx.x * blockDim.x + threadIdx.x;       // 392*32*16 = 200704
    if (t >= KCH * 32 * 16) return;
    int e  = t & 15;
    int L  = (t >> 4) & 31;
    int kc = t >> 9;
    int g  = L >> 4;
    int K  = kc * 32 + kmap(e, g);
    int n  = L & 15;
    float v = (n < 2) ? clsw[n * KDIM + K]
            : (n < 10) ? bboxw[(n - 2) * KDIM + K] : 0.f;
    wfragF[t] = (_Float16)v;
}

// ---------------------------------------------------------------------------
// Head GEMM: x(2048 x 12544) f16 @ W^T(12544 x 16-padded) via WMMA.
// One wave per 16-row tile, 392 K-steps, prefetching A 8 chunks ahead.
// ---------------------------------------------------------------------------
__global__ void fc_wmma_kernel(const _Float16* __restrict__ xmat,
                               const _Float16* __restrict__ wfragF,
                               const float* __restrict__ clsb,
                               const float* __restrict__ bboxb,
                               float* __restrict__ out) {
    int wave = (blockIdx.x * blockDim.x + threadIdx.x) >> 5;  // 128 waves
    int lane = threadIdx.x & 31;
    int g = lane >> 4;
    int m = lane & 15;
    const _Float16* arow = xmat + (size_t)(wave * 16 + m) * KDIM;

    v8f acc = {};
    #pragma unroll 4
    for (int kc = 0; kc < KCH; ++kc) {
        int k0 = kc * 32;
        half8 lo = *(const half8*)(arow + k0 + g * 8);
        half8 hi = *(const half8*)(arow + k0 + 16 + g * 8);
        __builtin_prefetch(arow + k0 + 256, 0, 0);       // A chunk 8 K-steps ahead
        v16h a = __builtin_shufflevector(lo, hi, 0,1,2,3,4,5,6,7,8,9,10,11,12,13,14,15);
        v16h bf = *(const v16h*)(wfragF + ((size_t)kc * 32 + lane) * 16);
        acc = __builtin_amdgcn_wmma_f32_16x16x32_f16(
            false, a, false, bf, (short)0, acc, false, false);
    }
    int n = m;                       // D: N = lane&15, M = v + 8*g
    if (n < 10) {
        float bias = (n < 2) ? clsb[n] : bboxb[n - 2];
        #pragma unroll
        for (int v = 0; v < 8; ++v) {
            int row = wave * 16 + v + 8 * g;
            out[row * 10 + n] = acc[v] + bias;
        }
    }
}

// ---------------------------------------------------------------------------
extern "C" void kernel_launch(void* const* d_in, const int* in_sizes, int n_in,
                              void* d_out, int out_size, void* d_ws, size_t ws_size,
                              hipStream_t stream) {
    (void)in_sizes; (void)n_in; (void)out_size; (void)ws_size;

    const float* images   = (const float*)d_in[0];
    const float* proposals= (const float*)d_in[1];
    const float* gt_boxes = (const float*)d_in[2];
    const float* conv1_w  = (const float*)d_in[3];
    const float* conv1_b  = (const float*)d_in[4];
    const float* conv2_w  = (const float*)d_in[5];
    const float* conv2_b  = (const float*)d_in[6];
    const float* cls_w    = (const float*)d_in[7];
    const float* cls_b    = (const float*)d_in[8];
    const float* bbox_w   = (const float*)d_in[9];
    const float* bbox_b   = (const float*)d_in[10];
    float* out = (float*)d_out;

    // workspace carve-up (256B aligned)
    auto align256 = [](size_t x) { return (x + 255) & ~(size_t)255; };
    char* ws = (char*)d_ws;
    size_t o = 0;
    _Float16* f1h    = (_Float16*)(ws + o); o = align256(o + (size_t)B_*H1*H1*C1*2);        // 8.39 MB
    _Float16* wfrag2 = (_Float16*)(ws + o); o = align256(o + (size_t)18*16*32*16*2);        // 288 KB
    float*    f2     = (float*)   (ws + o); o = align256(o + (size_t)B_*C2*H2*H2*4);        // 16.8 MB
    float*    rois   = (float*)   (ws + o); o = align256(o + (size_t)B_*NROI*4*4);          // 32 KB
    _Float16* xmat   = (_Float16*)(ws + o); o = align256(o + (size_t)B_*NROI*KDIM*2);       // 51.4 MB
    _Float16* wfragF = (_Float16*)(ws + o); o = align256(o + (size_t)KCH*32*16*2);          // 401 KB

    // 1) conv1 (direct, writes NHWC f16 activations)
    {
        int threads = B_ * C1 * H1 * H1;  // 4194304
        conv1_kernel<<<threads / 256, 256, 0, stream>>>(images, conv1_w, conv1_b, f1h);
    }
    // 2) weight pre-swizzles
    pack2_kernel<<<(18*16*32*16 + 255)/256, 256, 0, stream>>>(conv2_w, wfrag2);
    packF_kernel<<<(KCH*32*16 + 255)/256, 256, 0, stream>>>(cls_w, bbox_w, wfragF);
    // 3) match + sample (one block per batch)
    match_kernel<<<B_, 256, 0, stream>>>(gt_boxes, proposals, rois);
    // 4) conv2 implicit-GEMM WMMA with async-to-LDS A staging:
    //    grid = 4 batches * 256 M-tiles * 2 N-tile-groups
    conv2_wmma_kernel<<<B_ * 256 * 2, 256, 0, stream>>>(f1h, wfrag2, conv2_b, f2);
    // 5) ROI-align -> x matrix (f16)
    {
        int threads = B_ * NROI * C2 * 49;  // 25690112
        roi_kernel<<<threads / 256, 256, 0, stream>>>(f2, rois, xmat);
    }
    // 6) head GEMM WMMA: 128 waves = 16 blocks x 8 waves
    fc_wmma_kernel<<<16, 256, 0, stream>>>(xmat, wfragF, cls_b, bbox_b, out);
}
